// WindowAttention_60825326846683
// MI455X (gfx1250) — matrix operand order repaired and checked by reading, not compiled, not run on hardware
//
#include <hip/hip_runtime.h>

// ---------------------------------------------------------------------------
// Swin window attention on gfx1250 (wave32, WMMA 16x16x32 f16 -> f32 acc)
// One workgroup (256 threads = 8 wave32) per window (4096 windows).
// A/B fragments are cached in registers across the output tiles a wave owns.
// ---------------------------------------------------------------------------

#define NTOK   49          // tokens per window
#define NPAD   64          // padded to 4 WMMA row-tiles
#define DIMC   128         // channels
#define NHEAD  4
#define HDIM   32
#define NWIN   64          // windows per image (mask dim)
#define QSCALE 0.17677669529663687f   // 32^-0.5

// LDS strides (elements) -- row starts 16B aligned, bank-spread.
#define XS_LD  136   // f16: x tile / O tile, 64 x 128
#define WS_LD  136   // f16: staged weights (qkv_w 384x128, proj_w 128x128)
#define QK_LD  392   // f16: qkv output, 64 x 384
#define SB_LD  68    // f32: scores, per head 64 x 64
#define PR_LD  72    // f16: probs,  per head 64 x 64

// LDS byte offsets (all multiples of 16)
#define XS_OFF 0                         // 64*136*2      = 17408
#define QK_OFF 17408                     // 64*392*2      = 50176
#define U_OFF  67584                     // union: qkv weights (104448)
                                         //   OR scores (69632) + probs
                                         //   OR proj weights (in score area)
#define PB_OFF (U_OFF + 69632)           // probs 4*64*72*2 = 36864
#define SMEM_BYTES 174080

typedef _Float16 half8 __attribute__((ext_vector_type(8)));
typedef _Float16 v16h  __attribute__((ext_vector_type(16)));
typedef float    v8f   __attribute__((ext_vector_type(8)));

#define WMMA16(a, b, acc) \
  __builtin_amdgcn_wmma_f32_16x16x32_f16(false, (a), false, (b), (short)0, (acc), false, false)

// A fragment (16x32 f16, row-major source, stride ld):
// lane<16: row r, K in {hf*16 .. hf*16+15} (contiguous)
__device__ __forceinline__ v16h frag_a(const _Float16* p, int ld, int lane) {
  const int hf = lane >> 4, r = lane & 15;
  const _Float16* q = p + r * ld + hf * 16;
  half8 lo = *(const half8*)(q);
  half8 hi = *(const half8*)(q + 8);
  v16h a;
#pragma unroll
  for (int j = 0; j < 8; ++j) { a[j] = lo[j]; a[8 + j] = hi[j]; }
  return a;
}

// B fragment (32x16 f16) where memory holds B transposed:
// p[n*ld + k] == B[k][n]  ->  per-lane contiguous 16 halfs.
__device__ __forceinline__ v16h frag_b_cont(const _Float16* p, int ld, int lane) {
  const int hf = lane >> 4, c = lane & 15;
  const _Float16* q = p + c * ld + hf * 16;
  half8 lo = *(const half8*)(q);
  half8 hi = *(const half8*)(q + 8);
  v16h b;
#pragma unroll
  for (int j = 0; j < 8; ++j) { b[j] = lo[j]; b[8 + j] = hi[j]; }
  return b;
}

// B fragment (32x16 f16) where memory holds B directly: p[k*ld + n] == B[k][n].
__device__ __forceinline__ v16h frag_b_strided(const _Float16* p, int ld, int lane) {
  const int hf = lane >> 4, c = lane & 15;
  const _Float16* q = p + hf * 16 * ld + c;
  v16h b;
#pragma unroll
  for (int j = 0; j < 16; ++j) b[j] = q[j * ld];
  return b;
}

extern "C" __global__ __launch_bounds__(256)
void swin_window_attn(const float* __restrict__ xg,
                      const float* __restrict__ maskp,
                      const float* __restrict__ qkv_w,
                      const float* __restrict__ qkv_b,
                      const float* __restrict__ proj_w,
                      const float* __restrict__ proj_b,
                      const float* __restrict__ rtab,
                      const int*   __restrict__ ridx,
                      float* __restrict__ outg) {
  extern __shared__ __align__(16) char smem[];
  _Float16* xs    = (_Float16*)(smem + XS_OFF);   // x tile, later O tile
  _Float16* qkvs  = (_Float16*)(smem + QK_OFF);   // qkv output (q pre-scaled)
  _Float16* ws    = (_Float16*)(smem + U_OFF);    // staged qkv weights
  float*    sbuf  = (float*)   (smem + U_OFF);    // scores (after QKV phase)
  _Float16* pw    = (_Float16*)(smem + U_OFF);    // staged proj weights
  _Float16* probs = (_Float16*)(smem + PB_OFF);   // softmax probs

  const int tid  = threadIdx.x;
  const int wave = tid >> 5;
  const int lane = tid & 31;
  const int hf   = lane >> 4;
  const int c    = lane & 15;
  const int blk  = blockIdx.x;
  const int wIdx = blk & (NWIN - 1);              // window index within image

  const float* x = xg + (size_t)blk * NTOK * DIMC;

  // ---- Phase 1: stage x (f32 -> f16, zero-pad rows 49..63) and qkv_w ------
  // Also pre-warm proj_w toward L2/L0 for phase 5 (global_prefetch_b8).
  __builtin_prefetch(proj_w + tid * 64, 0, 1);
  for (int i4 = tid; i4 < NPAD * DIMC / 4; i4 += 256) {
    int m = i4 >> 5, col = (i4 & 31) * 4;
    float4 v = make_float4(0.f, 0.f, 0.f, 0.f);
    if (m < NTOK) v = ((const float4*)x)[i4];
    _Float16* d = xs + m * XS_LD + col;
    d[0] = (_Float16)v.x; d[1] = (_Float16)v.y;
    d[2] = (_Float16)v.z; d[3] = (_Float16)v.w;
  }
  for (int i4 = tid; i4 < 3 * DIMC * DIMC / 4; i4 += 256) {
    int o = i4 >> 5, col = (i4 & 31) * 4;
    float4 v = ((const float4*)qkv_w)[i4];
    _Float16* d = ws + o * WS_LD + col;
    d[0] = (_Float16)v.x; d[1] = (_Float16)v.y;
    d[2] = (_Float16)v.z; d[3] = (_Float16)v.w;
  }
  __syncthreads();

  // ---- Phase 2: QKV GEMM (64x384) = (64x128)(128x384), B = qkv_w^T --------
  // wave -> fixed mt (pairs of waves split the 24 nt tiles); A frags hoisted.
  {
    int mt  = wave >> 1;
    int ntb = (wave & 1) * 12;
    v16h aa[4];
#pragma unroll
    for (int k0 = 0; k0 < 4; ++k0)
      aa[k0] = frag_a(xs + mt * 16 * XS_LD + k0 * 32, XS_LD, lane);
    for (int ni = 0; ni < 12; ++ni) {
      int nt = ntb + ni;
      v8f acc = {};
#pragma unroll
      for (int k0 = 0; k0 < 4; ++k0) {
        v16h b = frag_b_cont(ws + nt * 16 * WS_LD + k0 * 32, WS_LD, lane);
        acc = WMMA16(aa[k0], b, acc);
      }
      int n0 = nt * 16 + c;
      float bias = qkv_b[n0];
      float scl = (n0 < DIMC) ? QSCALE : 1.0f;   // pre-scale q
      _Float16* drow = qkvs + (mt * 16 + hf * 8) * QK_LD + n0;
#pragma unroll
      for (int i = 0; i < 8; ++i)
        drow[i * QK_LD] = (_Float16)((acc[i] + bias) * scl);
    }
  }
  __syncthreads();

  // ---- Phase 3: S = Q K^T per head (K-dim = 32, single WMMA per tile) -----
  {
    int h = wave >> 1;
    int mtb = (wave & 1) * 2;
#pragma unroll
    for (int mi = 0; mi < 2; ++mi) {
      int mt = mtb + mi;
      v16h a = frag_a(qkvs + mt * 16 * QK_LD + h * HDIM, QK_LD, lane);
#pragma unroll
      for (int nt = 0; nt < 4; ++nt) {
        v16h b = frag_b_cont(qkvs + nt * 16 * QK_LD + DIMC + h * HDIM, QK_LD, lane);
        v8f acc = {};
        acc = WMMA16(a, b, acc);
        float* srow = sbuf + (h * NPAD + mt * 16 + hf * 8) * SB_LD + nt * 16 + c;
#pragma unroll
        for (int i = 0; i < 8; ++i) srow[i * SB_LD] = acc[i];
      }
    }
  }
  __syncthreads();

  // ---- Phase 4: softmax (+rel-pos bias, +shift mask), 4 heads x 64 rows ---
  {
    int h = tid >> 6, row = tid & 63;
    _Float16* prow = probs + (h * NPAD + row) * PR_LD;
    if (row < NTOK) {
      float* srow = sbuf + (h * NPAD + row) * SB_LD;
      const int* ri = ridx + row * NTOK;
      const float* mr = maskp + ((size_t)wIdx * NTOK + row) * NTOK;
      float mx = -1e30f;
      for (int n = 0; n < NTOK; ++n) {
        float v = srow[n] + rtab[ri[n] * NHEAD + h] + mr[n];
        srow[n] = v;
        mx = fmaxf(mx, v);
      }
      float sum = 0.f;
      for (int n = 0; n < NTOK; ++n) {
        float e = __expf(srow[n] - mx);
        srow[n] = e;
        sum += e;
      }
      float inv = 1.0f / sum;
      for (int n = 0; n < NTOK; ++n) prow[n] = (_Float16)(srow[n] * inv);
      for (int n = NTOK; n < NPAD; ++n) prow[n] = (_Float16)0.f;
    } else {
      for (int n = 0; n < NPAD; ++n) prow[n] = (_Float16)0.f;
    }
  }
  __syncthreads();

  // ---- Phase 5: O = P V per head (K=64 -> 2 WMMA steps); V frags hoisted;
  //               also stage proj_w into the freed score region -------------
  {
    int h = wave >> 1;
    int mtb = (wave & 1) * 2;
    v16h bb[4];                         // [dt*2 + k0step], strided V fragments
#pragma unroll
    for (int dt = 0; dt < 2; ++dt)
#pragma unroll
      for (int ks = 0; ks < 2; ++ks)
        bb[dt * 2 + ks] = frag_b_strided(
            qkvs + ks * 32 * QK_LD + 2 * DIMC + h * HDIM + dt * 16, QK_LD, lane);
#pragma unroll
    for (int mi = 0; mi < 2; ++mi) {
      int mt = mtb + mi;
      v16h a0 = frag_a(probs + (h * NPAD + mt * 16) * PR_LD, PR_LD, lane);
      v16h a1 = frag_a(probs + (h * NPAD + mt * 16) * PR_LD + 32, PR_LD, lane);
#pragma unroll
      for (int dt = 0; dt < 2; ++dt) {
        v8f acc = {};
        acc = WMMA16(a0, bb[dt * 2 + 0], acc);
        acc = WMMA16(a1, bb[dt * 2 + 1], acc);
        _Float16* drow = xs + (mt * 16 + hf * 8) * XS_LD + h * HDIM + dt * 16 + c;
#pragma unroll
        for (int i = 0; i < 8; ++i) drow[i * XS_LD] = (_Float16)acc[i];
      }
    }
    for (int i4 = tid; i4 < DIMC * DIMC / 4; i4 += 256) {
      int o = i4 >> 5, col = (i4 & 31) * 4;
      float4 v = ((const float4*)proj_w)[i4];
      _Float16* d = pw + o * WS_LD + col;
      d[0] = (_Float16)v.x; d[1] = (_Float16)v.y;
      d[2] = (_Float16)v.z; d[3] = (_Float16)v.w;
    }
  }
  __syncthreads();

  // ---- Phase 6: out = O proj_w^T + proj_b; A frags hoisted per wave -------
  float* out = outg + (size_t)blk * NTOK * DIMC;
  {
    int mt  = wave >> 1;
    int ntb = (wave & 1) * 4;
    v16h aa[4];
#pragma unroll
    for (int k0 = 0; k0 < 4; ++k0)
      aa[k0] = frag_a(xs + mt * 16 * XS_LD + k0 * 32, XS_LD, lane);
#pragma unroll
    for (int ni = 0; ni < 4; ++ni) {
      int nt = ntb + ni;
      v8f acc = {};
#pragma unroll
      for (int k0 = 0; k0 < 4; ++k0) {
        v16h b = frag_b_cont(pw + nt * 16 * WS_LD + k0 * 32, WS_LD, lane);
        acc = WMMA16(aa[k0], b, acc);
      }
      int n0 = nt * 16 + c;
      float pb = proj_b[n0];
#pragma unroll
      for (int i = 0; i < 8; ++i) {
        int m = mt * 16 + hf * 8 + i;
        if (m < NTOK) out[m * DIMC + n0] = acc[i] + pb;
      }
    }
  }
}

extern "C" void kernel_launch(void* const* d_in, const int* in_sizes, int n_in,
                              void* d_out, int out_size, void* d_ws, size_t ws_size,
                              hipStream_t stream) {
  (void)n_in; (void)out_size; (void)d_ws; (void)ws_size;
  const float* x      = (const float*)d_in[0];
  const float* mask   = (const float*)d_in[1];
  const float* qkv_w  = (const float*)d_in[2];
  const float* qkv_b  = (const float*)d_in[3];
  const float* proj_w = (const float*)d_in[4];
  const float* proj_b = (const float*)d_in[5];
  const float* rtab   = (const float*)d_in[6];
  const int*   ridx   = (const int*)d_in[7];
  float* out = (float*)d_out;

  int B = in_sizes[0] / (NTOK * DIMC);  // 4096 windows

  hipFuncSetAttribute((const void*)swin_window_attn,
                      hipFuncAttributeMaxDynamicSharedMemorySize, SMEM_BYTES);
  swin_window_attn<<<B, 256, SMEM_BYTES, stream>>>(
      x, mask, qkv_w, qkv_b, proj_w, proj_b, rtab, ridx, out);
}
